// AnnoCluster_89146341196335
// MI455X (gfx1250) — matrix operand order repaired and checked by reading, not compiled, not run on hardware
//
#include <hip/hip_runtime.h>
#include <hip/hip_bf16.h>
#include <math.h>

// ---------------------------------------------------------------------------
// Problem constants
// ---------------------------------------------------------------------------
#define BB 4096
#define DD 10000
#define HH 128
#define ZZ 32
#define KK 16
#define KMAIN 9984   // 312 full K=32 steps; 16-wide tail handled separately

// d_out layout (floats), reference tuple return order
#define OFF_EMEAN 0ULL
#define OFF_EDISP 40960000ULL
#define OFF_EPI   81920000ULL
#define OFF_QMEAN 122880000ULL
#define OFF_QDISP 163840000ULL
#define OFF_QPI   204800000ULL
#define OFF_ZE    245760000ULL
#define OFF_ZQ    245891072ULL
#define OFF_K     246022144ULL
#define OFF_MEAN  246026240ULL
#define OFF_PROB  246026241ULL

// workspace byte offsets
#define WS_HENC   0ULL         // 4096*128 f32  = 2 MiB
#define WS_HDECE  2097152ULL   // 4096*128 f16  = 1 MiB
#define WS_HDECQ  3145728ULL   // 4096*128 f16  = 1 MiB
#define WS_ACC    4194304ULL   // 1 f32

typedef __attribute__((ext_vector_type(16))) _Float16     v16h;
typedef __attribute__((ext_vector_type(8)))  float        v8f;
typedef __attribute__((ext_vector_type(4)))  unsigned int u32x4;
typedef __attribute__((ext_vector_type(4)))  int          i32x4;
typedef __attribute__((ext_vector_type(8)))  int          i32x8;

union Frag16 { v16h v; unsigned int u[8]; };

// ---------------------------------------------------------------------------
// WMMA fragment builders (ISA 7.12.2 layouts, wave32)
//
// A (16x32 f16, MxK): lane L -> M = L&15, half = L>>4.
//   VGPR r (pair a[2r],a[2r+1]) holds K = (r<4?0:16) + (half?8:0) + 2*(r&3), +1
// B (32x16 f16, KxN): lane L -> N = L&15, half = L>>4.
//   VGPR r holds K same as A: (r<4?0:16) + (half?8:0) + 2*(r&3), +1  (B staged transposed: Bt[n][k])
// C/D (16x16 f32): lane L -> N = L&15; VGPR r -> M = r + 8*(L>>4)
// ---------------------------------------------------------------------------
__device__ __forceinline__ v16h load_a_frag(const _Float16* At, int lda, int lane, int kb) {
    const int m = lane & 15, hf = lane >> 4;
    Frag16 f;
#pragma unroll
    for (int r = 0; r < 8; ++r) {
        const int k0 = ((r & 4) ? 16 : 0) + (hf ? 8 : 0) + 2 * (r & 3);
        f.u[r] = *(const unsigned int*)(At + m * lda + kb + k0);
    }
    return f.v;
}

__device__ __forceinline__ v16h load_b_frag(const _Float16* Bt, int ldb, int lane, int kb) {
    const int n = lane & 15, hf = lane >> 4;
    Frag16 f;
#pragma unroll
    for (int r = 0; r < 8; ++r) {
        f.u[r] = *(const unsigned int*)(Bt + n * ldb + kb + ((r & 4) ? 16 : 0) + (hf ? 8 : 0) + 2 * (r & 3));
    }
    return f.v;
}

// ---------------------------------------------------------------------------
// K1: encoder GEMM  h = relu(x[4096,10000] @ W1[10000,128] + b1) -> ws f32
// block = 256 thr (8 waves); block tile 16(M) x 128(N); wave w owns N-slice.
// Main K-loop (312 x K=32) has NO bounds checks; 16-wide tail is peeled.
// ---------------------------------------------------------------------------
__global__ void enc_gemm_kernel(const float* __restrict__ x,
                                const float* __restrict__ W1,
                                const float* __restrict__ b1,
                                float* __restrict__ hout) {
    __shared__ _Float16 At[16 * 34];
    __shared__ _Float16 Bt[128 * 34];

    const int t    = threadIdx.x;
    const int lane = t & 31;
    const int wv   = t >> 5;
    const int m0   = blockIdx.x * 16;

    v8f acc = {};

    // ---- main loop: guard-free steady state ----
    for (int kb = 0; kb < KMAIN; kb += 32) {
        if (t == 0) __builtin_prefetch(x + (size_t)m0 * DD + kb + 256, 0, 3);
#pragma unroll
        for (int i = 0; i < 2; ++i) {
            const int lin = t + i * 256;
            const int m = lin >> 5, kk = lin & 31;
            At[m * 34 + kk] = (_Float16)x[(size_t)(m0 + m) * DD + kb + kk];
        }
#pragma unroll
        for (int i = 0; i < 16; ++i) {
            const int lin = t + i * 256;
            const int k = lin >> 7, n = lin & 127;
            Bt[n * 34 + k] = (_Float16)W1[(size_t)(kb + k) * HH + n];
        }
        __syncthreads();
        const v16h a = load_a_frag(At, 34, lane, 0);
        const v16h b = load_b_frag(Bt + (wv * 16) * 34, 34, lane, 0);
        acc = __builtin_amdgcn_wmma_f32_16x16x32_f16(false, a, false, b, (short)0, acc, false, false);
        __syncthreads();
    }

    // ---- peeled K tail (cols 9984..9999, zero-padded to 32) ----
    {
#pragma unroll
        for (int i = 0; i < 2; ++i) {
            const int lin = t + i * 256;
            const int m = lin >> 5, kk = lin & 31, kg = KMAIN + kk;
            At[m * 34 + kk] = (kg < DD) ? (_Float16)x[(size_t)(m0 + m) * DD + kg] : (_Float16)0.f;
        }
#pragma unroll
        for (int i = 0; i < 16; ++i) {
            const int lin = t + i * 256;
            const int k = lin >> 7, n = lin & 127, kg = KMAIN + k;
            Bt[n * 34 + k] = (kg < DD) ? (_Float16)W1[(size_t)kg * HH + n] : (_Float16)0.f;
        }
        __syncthreads();
        const v16h a = load_a_frag(At, 34, lane, 0);
        const v16h b = load_b_frag(Bt + (wv * 16) * 34, 34, lane, 0);
        acc = __builtin_amdgcn_wmma_f32_16x16x32_f16(false, a, false, b, (short)0, acc, false, false);
    }

    const int n  = wv * 16 + (lane & 15);
    const int hf = lane >> 4;
    const float bn = b1[n];
#pragma unroll
    for (int r = 0; r < 8; ++r) {
        const int m = r + 8 * hf;
        float v = acc[r] + bn;
        hout[(size_t)(m0 + m) * HH + n] = v > 0.f ? v : 0.f;
    }
}

// ---------------------------------------------------------------------------
// K2: per-row latent + student-t assignment (one wave32 per row)
// ---------------------------------------------------------------------------
__global__ void assign_kernel(const float* __restrict__ h,
                              const float* __restrict__ Wmu,
                              const float* __restrict__ bmu,
                              const float* __restrict__ emb1,
                              const float* __restrict__ emb2,
                              const int*   __restrict__ subject,
                              float* __restrict__ out,
                              float* __restrict__ acc) {
    const int t    = threadIdx.x;
    const int lane = t & 31;
    const int row  = blockIdx.x * 8 + (t >> 5);

    const float* hr = h + (size_t)row * HH;
    float z = bmu[lane];
#pragma unroll 8
    for (int i = 0; i < HH; ++i) z += hr[i] * Wmu[i * ZZ + lane];
    out[OFF_ZE + (size_t)row * ZZ + lane] = z;

    const float* emb = (*subject == 0) ? emb1 : emb2;
    const bool valid = lane < KK;

    float d = 0.f;
#pragma unroll
    for (int i = 0; i < ZZ; ++i) {
        const float zi = __shfl(z, i);
        const float ev = valid ? emb[lane * ZZ + i] : 0.f;
        const float df = zi - ev;
        d += df * df;
    }
    const float dm = valid ? d : 0.f;

    float p = valid ? __expf(-5.5f * __logf(1.0f + dm * 0.1f)) : 0.f;
    float psum = p, dsum = dm;
#pragma unroll
    for (int off = 16; off; off >>= 1) {
        psum += __shfl_xor(psum, off);
        dsum += __shfl_xor(dsum, off);
    }
    if (valid) out[OFF_PROB + (size_t)row * KK + lane] = p / psum;
    if (lane == 0) atomicAdd(acc, dsum);

    float dd = valid ? d : 3.4e38f;
    int idx = lane;
#pragma unroll
    for (int off = 16; off; off >>= 1) {
        const float od = __shfl_xor(dd, off);
        const int   oi = __shfl_xor(idx, off);
        if (od < dd || (od == dd && oi < idx)) { dd = od; idx = oi; }
    }
    if (lane == 0) out[OFF_K + row] = (float)idx;
    out[OFF_ZQ + (size_t)row * ZZ + lane] = emb[idx * ZZ + lane];
}

// ---------------------------------------------------------------------------
// K3: decoder hidden  hdec = relu(z[4096,32] @ Wh[32,128] + bh) -> ws f16
// ---------------------------------------------------------------------------
__global__ void dec_hid_kernel(const float* __restrict__ z,
                               const float* __restrict__ Wh,
                               const float* __restrict__ bh,
                               _Float16* __restrict__ hout) {
    __shared__ _Float16 At[16 * 34];
    __shared__ _Float16 Bt[128 * 34];

    const int t    = threadIdx.x;
    const int lane = t & 31;
    const int wv   = t >> 5;
    const int m0   = blockIdx.x * 16;

#pragma unroll
    for (int i = 0; i < 2; ++i) {
        const int lin = t + i * 256;
        const int m = lin >> 5, kk = lin & 31;
        At[m * 34 + kk] = (_Float16)z[(size_t)(m0 + m) * ZZ + kk];
    }
#pragma unroll
    for (int i = 0; i < 16; ++i) {
        const int lin = t + i * 256;
        const int k = lin >> 7, n = lin & 127;
        Bt[n * 34 + k] = (_Float16)Wh[(size_t)k * HH + n];
    }
    __syncthreads();

    v8f acc = {};
    const v16h a = load_a_frag(At, 34, lane, 0);
    const v16h b = load_b_frag(Bt + (wv * 16) * 34, 34, lane, 0);
    acc = __builtin_amdgcn_wmma_f32_16x16x32_f16(false, a, false, b, (short)0, acc, false, false);

    const int n  = wv * 16 + (lane & 15);
    const int hf = lane >> 4;
    const float bn = bh[n];
#pragma unroll
    for (int r = 0; r < 8; ++r) {
        const int m = r + 8 * hf;
        float v = acc[r] + bn;
        hout[(size_t)(m0 + m) * HH + n] = (_Float16)(v > 0.f ? v : 0.f);
    }
}

// ---------------------------------------------------------------------------
// K4: heavy fused GEMM  out = act(hdec[4096,128] @ W[128,10000] + b)
// block = 128 thr (4 waves), block tile 64(M) x 16(N), full K staged once.
//
// The 64x128 f16 A panel is staged by the Tensor Data Mover:
//   2D tile, data_size=2B, tile 128 x 64, tensor_dim0_stride=128,
//   LDS padding: pad_interval=64 dwords (one row), pad_amount=2 dwords
//   -> LDS row stride 132 halves, identical to the manual layout.
// One wave issues the descriptor, waits TENSORcnt==0, then the block syncs.
// ---------------------------------------------------------------------------
template <int ACT>
__global__ void zinb_out_kernel(const _Float16* __restrict__ Hdec,
                                const float* __restrict__ W,
                                const float* __restrict__ bias,
                                float* __restrict__ out) {
    __shared__ _Float16 At[64 * 132];
    __shared__ _Float16 Bt[16 * 132];

    const int t    = threadIdx.x;
    const int lane = t & 31;
    const int wv   = t >> 5;
    const int n0   = blockIdx.x * 16;
    const int m0   = blockIdx.y * 64;

#if __has_builtin(__builtin_amdgcn_tensor_load_to_lds)
    if (wv == 0) {
        const unsigned long long ga =
            (unsigned long long)(uintptr_t)(Hdec + (size_t)m0 * HH);
        const unsigned int lds = (unsigned int)(uintptr_t)(void*)At; // addr[31:0] = LDS offset

        u32x4 g0;
        g0[0] = 1u;                                   // count=1, user mode
        g0[1] = lds;                                  // lds_addr
        g0[2] = (unsigned int)ga;                     // global_addr[31:0]
        g0[3] = (unsigned int)((ga >> 32) & 0x1FFFFFFu) | (2u << 30); // addr[56:32] | type=2

        i32x8 g1;
        g1[0] = (int)((1u << 16) |                    // data_size = 2B
                      (1u << 20) |                    // pad_enable
                      (5u << 22) |                    // pad_interval = 64 dwords
                      (1u << 25));                    // pad_amount   = 2 dwords
        g1[1] = (int)(128u << 16);                    // tensor_dim0 = 128 (lo16)
        g1[2] = (int)(4096u << 16);                   // tensor_dim1 = 4096 (lo16)
        g1[3] = (int)(128u << 16);                    // tile_dim0 = 128
        g1[4] = 64;                                   // tile_dim1 = 64 rows
        g1[5] = 128;                                  // tensor_dim0_stride = 128
        g1[6] = 0;
        g1[7] = 0;

        i32x4 gz = {};
#if defined(__clang_major__) && (__clang_major__ >= 23)
        i32x8 gz8 = {};
        __builtin_amdgcn_tensor_load_to_lds(g0, g1, gz, gz, gz8, 0);
#else
        __builtin_amdgcn_tensor_load_to_lds(g0, g1, gz, gz, 0);
#endif
        __builtin_amdgcn_s_wait_tensorcnt(0);
    }
#else
    // fallback: manual dword staging of the A panel
    {
        const unsigned int* H32 = (const unsigned int*)(Hdec + (size_t)m0 * HH);
#pragma unroll
        for (int i = 0; i < 32; ++i) {
            const int lin = t + i * 128;
            const int m = lin >> 6, k2 = lin & 63;
            *(unsigned int*)(At + m * 132 + 2 * k2) = H32[m * 64 + k2];
        }
    }
#endif

    // stage B transposed: Bt[n][k], 128 x 16 panel of W (f32 -> f16)
#pragma unroll
    for (int i = 0; i < 16; ++i) {
        const int lin = t + i * 128;          // 2048 elems
        const int k = lin >> 4, n = lin & 15;
        Bt[n * 132 + k] = (_Float16)W[(size_t)k * DD + n0 + n];
    }
    __syncthreads();

    v8f acc = {};
    const _Float16* Aw = At + (wv * 16) * 132;
#pragma unroll
    for (int kb = 0; kb < HH; kb += 32) {
        const v16h a = load_a_frag(Aw, 132, lane, kb);
        const v16h b = load_b_frag(Bt, 132, lane, kb);
        acc = __builtin_amdgcn_wmma_f32_16x16x32_f16(false, a, false, b, (short)0, acc, false, false);
    }

    const int n  = n0 + (lane & 15);
    const int hf = lane >> 4;
    const float bn = bias[n];
#pragma unroll
    for (int r = 0; r < 8; ++r) {
        const int m = m0 + wv * 16 + r + 8 * hf;
        float v = acc[r] + bn;
        float o;
        if (ACT == 0) {                      // MeanAct: exp(clip(v,-15,15))
            v = fminf(fmaxf(v, -15.f), 15.f);
            o = __expf(v);
        } else if (ACT == 1) {               // DispAct: softplus
            o = (v > 15.f) ? v : log1pf(__expf(v));
        } else {                             // sigmoid
            o = 1.f / (1.f + __expf(-v));
        }
        out[(size_t)m * DD + n] = o;
    }
}

// ---------------------------------------------------------------------------
__global__ void init_kernel(float* acc) { *acc = 0.f; }
__global__ void fin_kernel(const float* acc, float* out) {
    out[OFF_MEAN] = *acc * (1.0f / ((float)BB * (float)KK));
}

// ---------------------------------------------------------------------------
extern "C" void kernel_launch(void* const* d_in, const int* in_sizes, int n_in,
                              void* d_out, int out_size, void* d_ws, size_t ws_size,
                              hipStream_t stream) {
    const float* x       = (const float*)d_in[0];
    const int*   subject = (const int*)  d_in[1];
    const float* emb1    = (const float*)d_in[2];
    const float* emb2    = (const float*)d_in[3];
    const float* enc_W1  = (const float*)d_in[4];
    const float* enc_b1  = (const float*)d_in[5];
    const float* enc_Wmu = (const float*)d_in[6];
    const float* enc_bmu = (const float*)d_in[7];
    const float* e_Wh = (const float*)d_in[8];   const float* e_bh = (const float*)d_in[9];
    const float* e_Wm = (const float*)d_in[10];  const float* e_bm = (const float*)d_in[11];
    const float* e_Wd = (const float*)d_in[12];  const float* e_bd = (const float*)d_in[13];
    const float* e_Wp = (const float*)d_in[14];  const float* e_bp = (const float*)d_in[15];
    const float* q_Wh = (const float*)d_in[16];  const float* q_bh = (const float*)d_in[17];
    const float* q_Wm = (const float*)d_in[18];  const float* q_bm = (const float*)d_in[19];
    const float* q_Wd = (const float*)d_in[20];  const float* q_bd = (const float*)d_in[21];
    const float* q_Wp = (const float*)d_in[22];  const float* q_bp = (const float*)d_in[23];

    float* out = (float*)d_out;
    float*     h_enc  = (float*)    ((char*)d_ws + WS_HENC);
    _Float16*  hdec_e = (_Float16*) ((char*)d_ws + WS_HDECE);
    _Float16*  hdec_q = (_Float16*) ((char*)d_ws + WS_HDECQ);
    float*     acc    = (float*)    ((char*)d_ws + WS_ACC);

    init_kernel<<<1, 1, 0, stream>>>(acc);

    enc_gemm_kernel<<<BB / 16, 256, 0, stream>>>(x, enc_W1, enc_b1, h_enc);

    assign_kernel<<<BB / 8, 256, 0, stream>>>(h_enc, enc_Wmu, enc_bmu, emb1, emb2,
                                              subject, out, acc);
    fin_kernel<<<1, 1, 0, stream>>>(acc, out);

    dec_hid_kernel<<<BB / 16, 256, 0, stream>>>(out + OFF_ZE, e_Wh, e_bh, hdec_e);
    dec_hid_kernel<<<BB / 16, 256, 0, stream>>>(out + OFF_ZQ, q_Wh, q_bh, hdec_q);

    dim3 grid(DD / 16, BB / 64);
    zinb_out_kernel<0><<<grid, 128, 0, stream>>>(hdec_e, e_Wm, e_bm, out + OFF_EMEAN);
    zinb_out_kernel<1><<<grid, 128, 0, stream>>>(hdec_e, e_Wd, e_bd, out + OFF_EDISP);
    zinb_out_kernel<2><<<grid, 128, 0, stream>>>(hdec_e, e_Wp, e_bp, out + OFF_EPI);
    zinb_out_kernel<0><<<grid, 128, 0, stream>>>(hdec_q, q_Wm, q_bm, out + OFF_QMEAN);
    zinb_out_kernel<1><<<grid, 128, 0, stream>>>(hdec_q, q_Wd, q_bd, out + OFF_QDISP);
    zinb_out_kernel<2><<<grid, 128, 0, stream>>>(hdec_q, q_Wp, q_bp, out + OFF_QPI);
}
